// AttentionLayer_67370857005185
// MI455X (gfx1250) — matrix operand order repaired and checked by reading, not compiled, not run on hardware
//
#include <hip/hip_runtime.h>
#include <hip/hip_bf16.h>

#define HDIM 512
#define NHD  8
#define HD   64
#define BB   2
#define SS   4096
#define MM   (BB*SS)   // 8192 rows

typedef __attribute__((ext_vector_type(16))) __bf16 v16bf;
typedef __attribute__((ext_vector_type(8)))  __bf16 v8bf;
typedef __attribute__((ext_vector_type(8)))  float  v8f;
typedef __attribute__((ext_vector_type(4)))  float  v4f;
typedef __attribute__((ext_vector_type(4)))  unsigned int u32x4;
typedef __attribute__((ext_vector_type(8)))  int i32x8;
typedef __attribute__((ext_vector_type(4)))  int i32x4;

#define HAVE_TDM __has_builtin(__builtin_amdgcn_tensor_load_to_lds)

__device__ __forceinline__ v8f wmma_bf16(v16bf a, v16bf b, v8f c) {
  // (neg_a, A, neg_b, B, c_mod, C, reuse_a, reuse_b)
  return __builtin_amdgcn_wmma_f32_16x16x32_bf16(false, a, false, b, (short)0, c, false, false);
}

__device__ __forceinline__ v16bf pack16(v8bf lo, v8bf hi) {
  v16bf r;
#pragma unroll
  for (int i = 0; i < 8; ++i) { r[i] = lo[i]; r[8 + i] = hi[i]; }
  return r;
}

// ---------------------------------------------------------------------------
// Kernel 0: one-shot f32 -> bf16 cast of activations + weights.
// grid = (2048, 5), block 256; each thread converts 8 elements (b128 in/out).
// ---------------------------------------------------------------------------
__global__ __launch_bounds__(256) void cast_kernel(
    const float* __restrict__ X,  const float* __restrict__ Wq,
    const float* __restrict__ Wk, const float* __restrict__ Wv,
    const float* __restrict__ Wo,
    __bf16* __restrict__ Xb,  __bf16* __restrict__ Wqb,
    __bf16* __restrict__ Wkb, __bf16* __restrict__ Wvb,
    __bf16* __restrict__ Wob)
{
  const float* src; __bf16* dst; int n;
  switch (blockIdx.y) {
    case 0:  src = X;  dst = Xb;  n = MM * HDIM;   break;
    case 1:  src = Wq; dst = Wqb; n = HDIM * HDIM; break;
    case 2:  src = Wk; dst = Wkb; n = HDIM * HDIM; break;
    case 3:  src = Wv; dst = Wvb; n = HDIM * HDIM; break;
    default: src = Wo; dst = Wob; n = HDIM * HDIM; break;
  }
  const int i = (blockIdx.x * 256 + threadIdx.x) * 8;
  if (i < n) {
    v4f a = *(const v4f*)(src + i);
    v4f b = *(const v4f*)(src + i + 4);
    v8bf o;
#pragma unroll
    for (int k = 0; k < 4; ++k) { o[k] = (__bf16)a[k]; o[4 + k] = (__bf16)b[k]; }
    *(v8bf*)(dst + i) = o;
  }
}

// ---------------------------------------------------------------------------
// Kernel 1: fused QKV projection (all-bf16 loads).  Y = X @ W^T + b, stored
// bf16 head-major [B, NH, S, HD].  Q pre-scaled by 1/sqrt(HD).
// grid = (512, 3)  block = 256 (8 waves); each wave -> one 16x64 output tile.
// ---------------------------------------------------------------------------
__global__ __launch_bounds__(256) void qkv_kernel(
    const __bf16* __restrict__ Xb,
    const __bf16* __restrict__ Wqb, const float* __restrict__ bq,
    const __bf16* __restrict__ Wkb, const float* __restrict__ bk,
    const __bf16* __restrict__ Wvb, const float* __restrict__ bvp,
    __bf16* __restrict__ Qd, __bf16* __restrict__ Kd, __bf16* __restrict__ Vd)
{
  const int wave = threadIdx.x >> 5;
  const int lane = threadIdx.x & 31;
  const int hv   = lane >> 4;      // half-wave (0/1)
  const int mr   = lane & 15;
  const int gid  = blockIdx.x * 8 + wave;
  const int m0   = (gid >> 3) * 16;   // row tile
  const int n0   = (gid & 7) * 64;    // col tile

  const __bf16* W; const float* bias; __bf16* dst; float scale;
  if (blockIdx.y == 0)      { W = Wqb; bias = bq;  dst = Qd; scale = 0.125f; }
  else if (blockIdx.y == 1) { W = Wkb; bias = bk;  dst = Kd; scale = 1.0f;  }
  else                      { W = Wvb; bias = bvp; dst = Vd; scale = 1.0f;  }

  v8f acc[4];
#pragma unroll
  for (int t = 0; t < 4; ++t) acc[t] = 0;

  const __bf16* xrow = Xb + (size_t)(m0 + mr) * HDIM;
  for (int k0 = 0; k0 < HDIM; k0 += 32) {
    // A fragment: row m0+mr, runs [k0+8*hv, +8) and [k0+16+8*hv, +8)
    const __bf16* r = xrow + k0 + 8 * hv;
    v16bf a = pack16(*(const v8bf*)r, *(const v8bf*)(r + 16));
#pragma unroll
    for (int t = 0; t < 4; ++t) {
      // B fragment: column n0+t*16+mr of W^T == row of W, runs [k0+8*hv, +8) and [k0+16+8*hv, +8)
      const __bf16* wrow = W + (size_t)(n0 + t * 16 + mr) * HDIM + k0 + 8 * hv;
      v16bf b = pack16(*(const v8bf*)wrow, *(const v8bf*)(wrow + 16));
      acc[t] = wmma_bf16(a, b, acc[t]);
    }
  }

  const int bidx = (m0 >= SS) ? 1 : 0;
#pragma unroll
  for (int t = 0; t < 4; ++t) {
    const int n = n0 + t * 16 + mr;
    const int h = n >> 6, d = n & 63;
    const float bb = bias[n];
    __bf16* base = dst + ((size_t)(bidx * NHD + h) * SS) * HD + d;
#pragma unroll
    for (int e = 0; e < 8; ++e) {
      const int m = m0 + e + 8 * hv;
      const int s = m & (SS - 1);
      base[(size_t)s * HD] = (__bf16)((acc[t][e] + bb) * scale);
    }
  }
}

// ---------------------------------------------------------------------------
// Kernel 2: flash attention per head.  grid = (S/64, B*NH), block = 128.
// Each wave owns 16 query rows; K block DMA'd into LDS by the Tensor Data
// Mover (with hardware padding to the 72-element row pitch), V block staged
// transposed; online softmax; P relayouted via wave-private LDS.
// ---------------------------------------------------------------------------
__global__ __launch_bounds__(128) void attn_kernel(
    const __bf16* __restrict__ Qd, const __bf16* __restrict__ Kd,
    const __bf16* __restrict__ Vd, __bf16* __restrict__ Ctx)
{
  __shared__ __align__(16) __bf16 Kl[64][72];      // [key][d]
  __shared__ __align__(16) __bf16 Vt[64][72];      // [d][key]  (transposed)
  __shared__ __align__(16) __bf16 Pl[4][16][72];   // per-wave P: [m][key]

  const int tid  = threadIdx.x;
  const int wave = tid >> 5;
  const int lane = tid & 31;
  const int hv   = lane >> 4;
  const int mr   = lane & 15;
  const int bh   = blockIdx.y;
  const int s0   = blockIdx.x * 64 + wave * 16;
  const size_t headoff = (size_t)bh * SS * HD;

  // Q fragments stay resident: 2 chunks of k (d = 0..31, 32..63)
  const __bf16* Qp = Qd + headoff + (size_t)s0 * HD;
  v16bf qa[2];
#pragma unroll
  for (int c = 0; c < 2; ++c) {
    const __bf16* r = Qp + (size_t)mr * HD + c * 32 + 8 * hv;
    qa[c] = pack16(*(const v8bf*)r, *(const v8bf*)(r + 16));
  }

  v8f o[4];
#pragma unroll
  for (int t = 0; t < 4; ++t) o[t] = 0;
  float l[8], mx[8];
#pragma unroll
  for (int e = 0; e < 8; ++e) { l[e] = 0.0f; mx[e] = -1e30f; }

  const int NB   = SS / 64;
  const int srow = tid >> 1;          // staging row 0..63
  const int cb   = (tid & 1) * 32;    // staging col base

  for (int j = 0; j < NB; ++j) {
    const __bf16* Kp = Kd + headoff + (size_t)j * 64 * HD;
    const __bf16* Vp = Vd + headoff + (size_t)j * 64 * HD;

    __syncthreads();   // previous block fully consumed before restaging

#if HAVE_TDM
    if (wave == 0) {
      // Tensor DMA: 64x64 bf16 tile -> LDS, 32-DWORD rows padded by 4 DWORDs
      // to match the [64][72] layout.
      const unsigned long long ga = (unsigned long long)(uintptr_t)Kp;
      const unsigned lds = (unsigned)(uintptr_t)&Kl[0][0];
      u32x4 g0;
      g0[0] = 1u;                                    // count = 1 (valid D#)
      g0[1] = lds;                                   // lds_addr
      g0[2] = (unsigned)(ga & 0xFFFFFFFFu);          // global_addr lo
      g0[3] = (unsigned)((ga >> 32) & 0x1FFFFFFu)    // global_addr hi
              | (2u << 30);                          // type = 2 ("image")
      i32x8 g1;
      g1[0] = (1 << 16)            // data_size = 2 bytes
            | (1 << 20)            // pad_enable
            | (4 << 22)            // pad_interval: 32 DWORDs
            | (3 << 25);           // pad_amount:   4 DWORDs
      g1[1] = 64 << 16;            // tensor_dim0 = 64 (low 16 bits)
      g1[2] = 64 << 16;            // tensor_dim1 = 64 (low 16 bits)
      g1[3] = 64 << 16;            // tile_dim0 = 64
      g1[4] = 64;                  // tile_dim1 = 64
      g1[5] = 64;                  // tensor_dim0_stride = 64 (low 32 bits)
      g1[6] = 0;
      g1[7] = 0;
      i32x4 gz = {0, 0, 0, 0};
#if __clang_major__ >= 23
      i32x8 gz8 = {0, 0, 0, 0, 0, 0, 0, 0};
      __builtin_amdgcn_tensor_load_to_lds(g0, g1, gz, gz, gz8, 0);
#else
      __builtin_amdgcn_tensor_load_to_lds(g0, g1, gz, gz, 0);
#endif
    }
#endif

#pragma unroll
    for (int u = 0; u < 4; ++u) {
#if !HAVE_TDM
      *(v8bf*)&Kl[srow][cb + u * 8] = *(const v8bf*)(Kp + srow * HD + cb + u * 8);
#endif
      v8bf vv = *(const v8bf*)(Vp + srow * HD + cb + u * 8);
#pragma unroll
      for (int i = 0; i < 8; ++i) Vt[cb + u * 8 + i][srow] = vv[i];
    }

#if HAVE_TDM
    if (wave == 0) __builtin_amdgcn_s_wait_tensorcnt(0);
#endif
    __syncthreads();

    if (j + 1 < NB) {   // global_prefetch_b8 of next K/V block
      __builtin_prefetch(Kp + 64 * HD + tid * 32, 0, 0);
      __builtin_prefetch(Vp + 64 * HD + tid * 32, 0, 0);
    }

    // scores: 16x64 = 4 tiles, K-dim 64 = 2 chained wmma each
    v8f sc[4];
#pragma unroll
    for (int t = 0; t < 4; ++t) sc[t] = 0;
#pragma unroll
    for (int c = 0; c < 2; ++c) {
#pragma unroll
      for (int t = 0; t < 4; ++t) {
        const __bf16* kr = &Kl[t * 16 + mr][c * 32 + 8 * hv];
        v16bf b = pack16(*(const v8bf*)kr, *(const v8bf*)(kr + 16));
        sc[t] = wmma_bf16(qa[c], b, sc[t]);
      }
    }

    // online softmax (row stats per vector element; columns across 16 lanes)
    float corr[8];
#pragma unroll
    for (int e = 0; e < 8; ++e) {
      float sm = fmaxf(fmaxf(sc[0][e], sc[1][e]), fmaxf(sc[2][e], sc[3][e]));
      sm = fmaxf(sm, __shfl_xor(sm, 1));
      sm = fmaxf(sm, __shfl_xor(sm, 2));
      sm = fmaxf(sm, __shfl_xor(sm, 4));
      sm = fmaxf(sm, __shfl_xor(sm, 8));
      const float nm = fmaxf(mx[e], sm);
      corr[e] = __expf(mx[e] - nm);
      mx[e] = nm;
      float rs = 0.0f;
#pragma unroll
      for (int t = 0; t < 4; ++t) {
        float p = __expf(sc[t][e] - nm);
        sc[t][e] = p;
        rs += p;
      }
      rs += __shfl_xor(rs, 1);
      rs += __shfl_xor(rs, 2);
      rs += __shfl_xor(rs, 4);
      rs += __shfl_xor(rs, 8);
      l[e] = l[e] * corr[e] + rs;
    }
#pragma unroll
    for (int t = 0; t < 4; ++t)
#pragma unroll
      for (int e = 0; e < 8; ++e) o[t][e] *= corr[e];

    // P -> wave-private LDS (bf16), re-read as A fragments
    __bf16* Pw = &Pl[wave][0][0];
#pragma unroll
    for (int t = 0; t < 4; ++t)
#pragma unroll
      for (int e = 0; e < 8; ++e)
        Pw[(e + 8 * hv) * 72 + t * 16 + mr] = (__bf16)sc[t][e];

    // O += P @ V
#pragma unroll
    for (int c = 0; c < 2; ++c) {
      const __bf16* pr = Pw + mr * 72 + c * 32 + 8 * hv;
      v16bf pa = pack16(*(const v8bf*)pr, *(const v8bf*)(pr + 16));
#pragma unroll
      for (int t = 0; t < 4; ++t) {
        const __bf16* vr = &Vt[t * 16 + mr][c * 32 + 8 * hv];
        v16bf b = pack16(*(const v8bf*)vr, *(const v8bf*)(vr + 16));
        o[t] = wmma_bf16(pa, b, o[t]);
      }
    }
  }

  // epilogue: normalize and write ctx back to [B, S, H] (bf16)
  float inv[8];
#pragma unroll
  for (int e = 0; e < 8; ++e) inv[e] = 1.0f / l[e];
  const int b_ = bh >> 3, h_ = bh & 7;
#pragma unroll
  for (int t = 0; t < 4; ++t) {
    const int d = t * 16 + mr;
#pragma unroll
    for (int e = 0; e < 8; ++e) {
      const int sr = s0 + e + 8 * hv;
      Ctx[((size_t)b_ * SS + sr) * HDIM + h_ * HD + d] = (__bf16)(o[t][e] * inv[e]);
    }
  }
}

// ---------------------------------------------------------------------------
// Kernel 3: output projection.  out = ctx @ Wo^T + bo  (fp32 out).
// grid = 512, block = 256 (8 waves); each wave -> one 16x64 tile.
// ---------------------------------------------------------------------------
__global__ __launch_bounds__(256) void oproj_kernel(
    const __bf16* __restrict__ Ctx, const __bf16* __restrict__ Wob,
    const float* __restrict__ bo, float* __restrict__ Out)
{
  const int wave = threadIdx.x >> 5;
  const int lane = threadIdx.x & 31;
  const int hv   = lane >> 4;
  const int mr   = lane & 15;
  const int gid  = blockIdx.x * 8 + wave;
  const int m0   = (gid >> 3) * 16;
  const int n0   = (gid & 7) * 64;

  v8f acc[4];
#pragma unroll
  for (int t = 0; t < 4; ++t) acc[t] = 0;

  const __bf16* xrow = Ctx + (size_t)(m0 + mr) * HDIM;
  for (int k0 = 0; k0 < HDIM; k0 += 32) {
    const __bf16* r = xrow + k0 + 8 * hv;
    v16bf a = pack16(*(const v8bf*)r, *(const v8bf*)(r + 16));
#pragma unroll
    for (int t = 0; t < 4; ++t) {
      const __bf16* wrow = Wob + (size_t)(n0 + t * 16 + mr) * HDIM + k0 + 8 * hv;
      v16bf b = pack16(*(const v8bf*)wrow, *(const v8bf*)(wrow + 16));
      acc[t] = wmma_bf16(a, b, acc[t]);
    }
  }

#pragma unroll
  for (int t = 0; t < 4; ++t) {
    const int n = n0 + t * 16 + mr;
    const float bb = bo[n];
#pragma unroll
    for (int e = 0; e < 8; ++e) {
      const int m = m0 + e + 8 * hv;
      Out[(size_t)m * HDIM + n] = acc[t][e] + bb;
    }
  }
}

// ---------------------------------------------------------------------------
extern "C" void kernel_launch(void* const* d_in, const int* in_sizes, int n_in,
                              void* d_out, int out_size, void* d_ws, size_t ws_size,
                              hipStream_t stream)
{
  const float* X  = (const float*)d_in[0];
  const float* Wq = (const float*)d_in[1];
  const float* bq = (const float*)d_in[2];
  const float* Wk = (const float*)d_in[3];
  const float* bk = (const float*)d_in[4];
  const float* Wv = (const float*)d_in[5];
  const float* bv = (const float*)d_in[6];
  const float* Wo = (const float*)d_in[7];
  const float* bo = (const float*)d_in[8];
  float* Out = (float*)d_out;

  const size_t ME = (size_t)MM * HDIM;     // 4,194,304 elements
  const size_t WE = (size_t)HDIM * HDIM;   //   262,144 elements
  __bf16* Xb  = (__bf16*)d_ws;
  __bf16* Wqb = Xb  + ME;
  __bf16* Wkb = Wqb + WE;
  __bf16* Wvb = Wkb + WE;
  __bf16* Wob = Wvb + WE;
  __bf16* Qd  = Wob + WE;
  __bf16* Kd  = Qd  + ME;
  __bf16* Vd  = Kd  + ME;
  __bf16* Ctx = Vd  + ME;

  dim3 g0(ME / (256 * 8), 5);
  cast_kernel<<<g0, 256, 0, stream>>>(X, Wq, Wk, Wv, Wo, Xb, Wqb, Wkb, Wvb, Wob);

  dim3 g1(512, 3);
  qkv_kernel<<<g1, 256, 0, stream>>>(Xb, Wqb, bq, Wkb, bk, Wvb, bv, Qd, Kd, Vd);

  dim3 g2(SS / 64, BB * NHD);
  attn_kernel<<<g2, 128, 0, stream>>>(Qd, Kd, Vd, Ctx);

  oproj_kernel<<<512, 256, 0, stream>>>(Ctx, Wob, bo, Out);
}
